// AttMultiHeadLoc_87832081203821
// MI455X (gfx1250) — matrix operand order repaired and checked by reading, not compiled, not run on hardware
//
#include <hip/hip_runtime.h>
#include <hip/hip_bf16.h>

typedef __attribute__((ext_vector_type(16))) _Float16 v16h;
typedef __attribute__((ext_vector_type(8)))  float    v8f;

// Problem constants
#define B_   32
#define T_   2048
#define E_   1024
#define D_   1024
#define H_   4
#define DK_  512
#define DV_  512
#define C_   10
#define FILT_ 100
#define KW_  201
#define SCALING_ 2.0f

// ---------------------------------------------------------------------------
// zero a float buffer
__global__ void zero_kernel(float* __restrict__ p, int n) {
    int i = blockIdx.x * blockDim.x + threadIdx.x;
    if (i < n) p[i] = 0.f;
}

// ---------------------------------------------------------------------------
// Wk (H,E,DK) f32  ->  wkT (H,DK,E) f16   (LDS-tiled transpose + convert)
__global__ __launch_bounds__(256) void transpose_wk_kernel(
        const float* __restrict__ Wk, _Float16* __restrict__ wkT) {
    __shared__ float tile[32][33];
    int bid = blockIdx.x;
    int dkt = bid & 15;             // DK/32 = 16
    int et  = (bid >> 4) & 31;      // E/32  = 32
    int h   = bid >> 9;             // H = 4
    int e0 = et * 32, dk0 = dkt * 32;
    int tid = threadIdx.x;
    for (int rr = 0; rr < 4; ++rr) {
        int er = (tid >> 5) + rr * 8;
        int dc = tid & 31;
        tile[er][dc] = Wk[((size_t)(h * E_) + e0 + er) * DK_ + dk0 + dc];
    }
    __syncthreads();
    for (int rr = 0; rr < 4; ++rr) {
        int dr = (tid >> 5) + rr * 8;
        int ec = tid & 31;
        wkT[((size_t)(h * DK_) + dk0 + dr) * E_ + e0 + ec] = (_Float16)tile[ec][dr];
    }
}

// ---------------------------------------------------------------------------
// q[b,h,dk] = dec_z[b,:] . Wq[h,:,dk] + bq[h,dk]
__global__ __launch_bounds__(256) void q_kernel(
        const float* __restrict__ dec_z, const float* __restrict__ Wq,
        const float* __restrict__ bq, float* __restrict__ qv) {
    int b = blockIdx.x >> 2, h = blockIdx.x & 3;
    int tid = threadIdx.x;
    __shared__ float zS[D_];
    for (int i = tid; i < D_; i += 256) zS[i] = dec_z[b * D_ + i];
    __syncthreads();
    float a0 = bq[h * DK_ + tid];
    float a1 = bq[h * DK_ + tid + 256];
    for (int e = 0; e < D_; ++e) {
        const float* wr = Wq + ((size_t)(h * D_ + e)) * DK_;
        float z = zS[e];
        a0 += z * wr[tid];
        a1 += z * wr[tid + 256];
    }
    qv[(b * H_ + h) * DK_ + tid]       = a0;
    qv[(b * H_ + h) * DK_ + tid + 256] = a1;
}

// ---------------------------------------------------------------------------
// grouped location conv: conv_out[b,h,c,t] = sum_j att_prev[b,h,t+j-FILT]*conv_w[h,c,j]
__global__ __launch_bounds__(256) void conv_kernel(
        const float* __restrict__ att_prev, const float* __restrict__ conv_w,
        float* __restrict__ conv_out) {
    int b = blockIdx.x >> 2, h = blockIdx.x & 3;
    int tid = threadIdx.x;
    __shared__ float ap[T_];
    __shared__ float cw[C_ * KW_];
    for (int i = tid; i < T_; i += 256) ap[i] = att_prev[((size_t)(b * H_ + h)) * T_ + i];
    for (int i = tid; i < C_ * KW_; i += 256) cw[i] = conv_w[(size_t)h * C_ * KW_ + i];
    __syncthreads();
    for (int o = tid; o < C_ * T_; o += 256) {
        int c = o >> 11;           // /T_
        int t = o & (T_ - 1);
        int jlo = (t < FILT_) ? (FILT_ - t) : 0;
        int jhi = (t > T_ - 1 - FILT_) ? (T_ - 1 - t + FILT_ + 1) : KW_;
        const float* cwc = &cw[c * KW_];
        float acc = 0.f;
        for (int j = jlo; j < jhi; ++j) acc += ap[t - FILT_ + j] * cwc[j];
        conv_out[(((size_t)(b * H_ + h)) * C_ + c) * T_ + t] = acc;
    }
}

// ---------------------------------------------------------------------------
// Fused score kernel: for a 32-row t tile of (b,h):
//   S[t,dk] = (enc[b,t,:]f16) x (Wk[h,:,dk]f16)  via v_wmma_f32_16x16x32_f16
//   e[b,h,t] = sum_dk tanh(S + loc + q) * gw    (+ gb added here)
// 8 waves, each wave owns M=32 x N=64: 8 accumulators, B fragments reused
// across both M-subtiles (8 WMMA per 12 ds_load_b128 per K-chunk).
#define APITCH 40   // f16 pitch, 80B rows: 16B aligned, conflict-free frag reads
__global__ __launch_bounds__(256) void escore_kernel(
        const float* __restrict__ enc, const _Float16* __restrict__ wkT,
        const float* __restrict__ qv, const float* __restrict__ conv_out,
        const float* __restrict__ Watt, const float* __restrict__ gw,
        const float* __restrict__ gb, float* __restrict__ e_out) {
    __shared__ _Float16 As[32 * APITCH];
    __shared__ _Float16 Bs[DK_ * APITCH];
    __shared__ float convS[C_ * 32];
    __shared__ float esum[32];

    int bid   = blockIdx.x;
    int ttile = bid & 63;           // T/32 = 64
    int h     = (bid >> 6) & 3;
    int b     = bid >> 8;
    int t0    = ttile * 32;
    int tid   = threadIdx.x;
    int lane  = tid & 31;
    int wave  = tid >> 5;           // 0..7
    int nbase = wave * 64;

    if (tid < 32) esum[tid] = 0.f;
    for (int i = tid; i < C_ * 32; i += 256) {
        int c = i >> 5, m = i & 31;
        convS[c * 32 + m] = conv_out[(((size_t)(b * H_ + h)) * C_ + c) * T_ + t0 + m];
    }

    v8f acc[2][4] = {};             // [m-subtile][n-tile]

    const float*    encb = enc + ((size_t)b * T_ + t0) * E_;
    const _Float16* wkh  = wkT + (size_t)h * DK_ * E_;

    for (int kc = 0; kc < E_ / 32; ++kc) {
        int e0 = kc * 32;
        __syncthreads();
        // --- stage A tile (32 t x 32 e) f32 -> f16 : 4 elems/thread
        {
            int m  = tid >> 3;              // 0..31
            int kk = (tid & 7) * 4;         // 0..28
            float4 f = *(const float4*)(encb + (size_t)m * E_ + e0 + kk);
            union { _Float16 hh[4]; int2 v; } u;
            u.hh[0] = (_Float16)f.x; u.hh[1] = (_Float16)f.y;
            u.hh[2] = (_Float16)f.z; u.hh[3] = (_Float16)f.w;
            *(int2*)&As[m * APITCH + kk] = u.v;
        }
        // --- stage B tile (512 dk x 32 e) f16, K-major in wkT: 8 int4/thread
        {
            int j    = tid & 3;             // 4x 16B chunks per row
            int nrow = tid >> 2;            // 64 rows/pass
            for (int pass = 0; pass < 8; ++pass, nrow += 64) {
                int4 v = *(const int4*)(wkh + (size_t)nrow * E_ + e0 + j * 8);
                *(int4*)&Bs[nrow * APITCH + j * 8] = v;
            }
        }
        __syncthreads();
        // --- fragments per ISA 16-bit layouts
        union { v16h v; int4 q[2]; } afrag[2];
        int k0a = (lane < 16) ? 0 : 8;
        for (int ms = 0; ms < 2; ++ms) {
            int m = (lane & 15) + ms * 16;
            afrag[ms].q[0] = *(const int4*)&As[m * APITCH + k0a];       // K k0a..+7
            afrag[ms].q[1] = *(const int4*)&As[m * APITCH + 16 + k0a];  // K 16+k0a..
        }
        int k0b = (lane < 16) ? 0 : 8;
        for (int nt = 0; nt < 4; ++nt) {
            int n = nbase + nt * 16 + (lane & 15);
            union { v16h v; int4 q[2]; } bfrag;
            bfrag.q[0] = *(const int4*)&Bs[n * APITCH + k0b];
            bfrag.q[1] = *(const int4*)&Bs[n * APITCH + k0b + 16];
            acc[0][nt] = __builtin_amdgcn_wmma_f32_16x16x32_f16(
                false, afrag[0].v, false, bfrag.v, (short)0, acc[0][nt], false, false);
            acc[1][nt] = __builtin_amdgcn_wmma_f32_16x16x32_f16(
                false, afrag[1].v, false, bfrag.v, (short)0, acc[1][nt], false, false);
        }
    }

    // --- fused epilogue: +loc +q, tanh, *gw, reduce over dk
    float qn[4], gwn[4], wt[4][C_];
    for (int nt = 0; nt < 4; ++nt) {
        int n = nbase + nt * 16 + (lane & 15);
        qn[nt]  = qv[(b * H_ + h) * DK_ + n];
        gwn[nt] = gw[h * DK_ + n];
        for (int c = 0; c < C_; ++c) wt[nt][c] = Watt[((h * C_ + c)) * (size_t)DK_ + n];
    }
    int hi = (lane >= 16) ? 8 : 0;
    for (int ms = 0; ms < 2; ++ms) {
        for (int r = 0; r < 8; ++r) {
            int m = ms * 16 + hi + r;
            float cv[C_];
            for (int c = 0; c < C_; ++c) cv[c] = convS[c * 32 + m];
            float p = 0.f;
            for (int nt = 0; nt < 4; ++nt) {
                float loc = 0.f;
                for (int c = 0; c < C_; ++c) loc += cv[c] * wt[nt][c];
                float val = tanhf(acc[ms][nt][r] + loc + qn[nt]);
                p += val * gwn[nt];
            }
            p += __shfl_xor(p, 1, 32);
            p += __shfl_xor(p, 2, 32);
            p += __shfl_xor(p, 4, 32);
            p += __shfl_xor(p, 8, 32);
            if ((lane & 15) == 0) atomicAdd(&esum[m], p);
        }
    }
    __syncthreads();
    if (tid < 32)
        e_out[((size_t)(b * H_ + h)) * T_ + t0 + tid] = esum[tid] + gb[h];
}

// ---------------------------------------------------------------------------
// masked softmax over T with x2 scaling; writes w (B,H,T) into d_out region
__global__ __launch_bounds__(256) void softmax_kernel(
        const float* __restrict__ e_ws, const int* __restrict__ lens,
        float* __restrict__ wout) {
    int b = blockIdx.x >> 2, h = blockIdx.x & 3;
    int tid = threadIdx.x;
    int len = lens[b];
    const float* ep = e_ws + ((size_t)(b * H_ + h)) * T_;
    __shared__ float red[256];
    float m = -3.4e38f;
    for (int t = tid; t < T_; t += 256)
        if (t < len) m = fmaxf(m, SCALING_ * ep[t]);
    red[tid] = m; __syncthreads();
    for (int s = 128; s > 0; s >>= 1) {
        if (tid < s) red[tid] = fmaxf(red[tid], red[tid + s]);
        __syncthreads();
    }
    float mx = red[0]; __syncthreads();
    float sacc = 0.f;
    for (int t = tid; t < T_; t += 256)
        if (t < len) sacc += expf(SCALING_ * ep[t] - mx);
    red[tid] = sacc; __syncthreads();
    for (int s = 128; s > 0; s >>= 1) {
        if (tid < s) red[tid] += red[tid + s];
        __syncthreads();
    }
    float inv = 1.f / red[0];
    for (int t = tid; t < T_; t += 256) {
        float w = (t < len) ? expf(SCALING_ * ep[t] - mx) * inv : 0.f;
        wout[((size_t)(b * H_ + h)) * T_ + t] = w;
    }
}

// ---------------------------------------------------------------------------
// u[b,h,e] = sum_t w[b,h,t] * enc[b,t,e]   (t split 8 ways, f32 atomics)
__global__ __launch_bounds__(256) void u_kernel(
        const float* __restrict__ enc, const float* __restrict__ wglob,
        float* __restrict__ u) {
    int bid = blockIdx.x;
    int tc = bid & 7, ec = (bid >> 3) & 3, b = bid >> 5;
    int tid = threadIdx.x;
    int e  = ec * 256 + tid;
    int t0 = tc * 256;
    __shared__ float wS[H_][256];
    for (int i = tid; i < H_ * 256; i += 256) {
        int h = i >> 8, tt = i & 255;
        wS[h][tt] = wglob[((size_t)(b * H_ + h)) * T_ + t0 + tt];
    }
    __syncthreads();
    float a0 = 0.f, a1 = 0.f, a2 = 0.f, a3 = 0.f;
    const float* ep = enc + ((size_t)b * T_ + t0) * E_ + e;
    for (int tt = 0; tt < 256; ++tt) {
        float ev = ep[(size_t)tt * E_];
        a0 += wS[0][tt] * ev; a1 += wS[1][tt] * ev;
        a2 += wS[2][tt] * ev; a3 += wS[3][tt] * ev;
    }
    atomicAdd(&u[(b * H_ + 0) * E_ + e], a0);
    atomicAdd(&u[(b * H_ + 1) * E_ + e], a1);
    atomicAdd(&u[(b * H_ + 2) * E_ + e], a2);
    atomicAdd(&u[(b * H_ + 3) * E_ + e], a3);
}

// ---------------------------------------------------------------------------
// chead[b,h,dv] = sum_e u[b,h,e] * Wv[h,e,dv]
__global__ __launch_bounds__(256) void chead_kernel(
        const float* __restrict__ u, const float* __restrict__ Wv,
        float* __restrict__ chead) {
    int b = blockIdx.x >> 2, h = blockIdx.x & 3;
    int tid = threadIdx.x;
    __shared__ float uS[E_];
    for (int i = tid; i < E_; i += 256) uS[i] = u[(b * H_ + h) * E_ + i];
    __syncthreads();
    float a0 = 0.f, a1 = 0.f;
    for (int e = 0; e < E_; ++e) {
        const float* wr = Wv + ((size_t)(h * E_ + e)) * DV_;
        float uv = uS[e];
        a0 += uv * wr[tid];
        a1 += uv * wr[tid + 256];
    }
    chead[(b * H_ + h) * DV_ + tid]       = a0;
    chead[(b * H_ + h) * DV_ + tid + 256] = a1;
}

// ---------------------------------------------------------------------------
// c[b,eo] = sum_i chead[b,i] * Wo[i,eo]   (i over H*DV = 2048)
__global__ __launch_bounds__(256) void out_kernel(
        const float* __restrict__ chead, const float* __restrict__ Wo,
        float* __restrict__ out) {
    int b = blockIdx.x;
    int tid = threadIdx.x;
    __shared__ float cS[H_ * DV_];
    for (int i = tid; i < H_ * DV_; i += 256) cS[i] = chead[(size_t)b * H_ * DV_ + i];
    __syncthreads();
    float a0 = 0.f, a1 = 0.f, a2 = 0.f, a3 = 0.f;
    for (int i = 0; i < H_ * DV_; ++i) {
        const float* wr = Wo + (size_t)i * E_;
        float cv = cS[i];
        a0 += cv * wr[tid];
        a1 += cv * wr[tid + 256];
        a2 += cv * wr[tid + 512];
        a3 += cv * wr[tid + 768];
    }
    out[(size_t)b * E_ + tid]       = a0;
    out[(size_t)b * E_ + tid + 256] = a1;
    out[(size_t)b * E_ + tid + 512] = a2;
    out[(size_t)b * E_ + tid + 768] = a3;
}

// ---------------------------------------------------------------------------
extern "C" void kernel_launch(void* const* d_in, const int* in_sizes, int n_in,
                              void* d_out, int out_size, void* d_ws, size_t ws_size,
                              hipStream_t stream) {
    (void)in_sizes; (void)n_in; (void)out_size; (void)ws_size;
    const float* enc      = (const float*)d_in[0];
    const int*   lens     = (const int*)  d_in[1];
    const float* dec_z    = (const float*)d_in[2];
    const float* att_prev = (const float*)d_in[3];
    const float* Wq       = (const float*)d_in[4];
    const float* bq       = (const float*)d_in[5];
    const float* Wk       = (const float*)d_in[6];
    const float* Wv       = (const float*)d_in[7];
    const float* gw       = (const float*)d_in[8];
    const float* gb       = (const float*)d_in[9];
    const float* conv_w   = (const float*)d_in[10];
    const float* Watt     = (const float*)d_in[11];
    const float* Wo       = (const float*)d_in[12];
    float* out = (float*)d_out;

    // workspace layout (bytes)
    char* ws = (char*)d_ws;
    _Float16* wkT   = (_Float16*)(ws + 0);          //  4,194,304 B  (H,DK,E) f16
    float*    qv    = (float*)(ws + 4194304);       //    262,144 B  (B,H,DK)
    float*    convo = (float*)(ws + 4456448);       // 10,485,760 B  (B,H,C,T)
    float*    e_ws  = (float*)(ws + 14942208);      //  1,048,576 B  (B,H,T)
    float*    u     = (float*)(ws + 15990784);      //    524,288 B  (B,H,E)
    float*    chead = (float*)(ws + 16515072);      //    262,144 B  (B,H,DV)

    float* w_out = out + B_ * E_;                   // w region of d_out

    zero_kernel<<<(B_ * H_ * E_ + 255) / 256, 256, 0, stream>>>(u, B_ * H_ * E_);
    transpose_wk_kernel<<<H_ * (E_ / 32) * (DK_ / 32), 256, 0, stream>>>(Wk, wkT);
    q_kernel<<<B_ * H_, 256, 0, stream>>>(dec_z, Wq, bq, qv);
    conv_kernel<<<B_ * H_, 256, 0, stream>>>(att_prev, conv_w, convo);
    escore_kernel<<<B_ * H_ * (T_ / 32), 256, 0, stream>>>(
        enc, wkT, qv, convo, Watt, gw, gb, e_ws);
    softmax_kernel<<<B_ * H_, 256, 0, stream>>>(e_ws, lens, w_out);
    u_kernel<<<B_ * 4 * 8, 256, 0, stream>>>(enc, w_out, u);
    chead_kernel<<<B_ * H_, 256, 0, stream>>>(u, Wv, chead);
    out_kernel<<<B_, 256, 0, stream>>>(chead, Wo, out);
}
